// PerceiverResampler_21663815041180
// MI455X (gfx1250) — matrix operand order repaired and checked by reading, not compiled, not run on hardware
//
#include <hip/hip_runtime.h>
#include <math.h>

// ---------------- problem constants ----------------
#define G_    32      // B*M = 8*4 batch groups
#define NS_   1024    // source tokens
#define NLTN_ 64      // latents
#define NCTX_ 1088    // NS_+NLTN_
#define D_    768
#define E_    768
#define H_    12
#define DH_   64
#define DFF_  3072
#define L_    6

typedef __bf16 bf16_t;
typedef __attribute__((ext_vector_type(16))) __bf16 v16bf;
typedef __attribute__((ext_vector_type(8)))  __bf16 v8bf;
typedef __attribute__((ext_vector_type(2)))  __bf16 v2bf;
typedef __attribute__((ext_vector_type(8)))  float  v8f;
typedef __attribute__((ext_vector_type(8)))  unsigned v8u;

__device__ __forceinline__ bf16_t f2bf(float x) { return (bf16_t)x; }

__device__ __forceinline__ unsigned pk_bf16(float x, float y) {
  v2bf t; t[0] = (bf16_t)x; t[1] = (bf16_t)y;
  return __builtin_bit_cast(unsigned, t);
}

// ---------------- WMMA helper (CDNA5 16x16x32 bf16, f32 accum) ----------------
__device__ __forceinline__ v8f wmma_bf16(v16bf a, v16bf b, v8f c) {
  return __builtin_amdgcn_wmma_f32_16x16x32_bf16(false, a, false, b, (short)0, c,
                                                 false, false);
}

// A fragment: 16(M) x 32(K) bf16 from row-major [rows, ldk].
// lane<16: row=row0+lane, K=k0..k0+15; lane>=16: row-16, K=k0+16..k0+31
__device__ __forceinline__ v16bf load_a_frag(const bf16_t* base, int row0, int ldk, int k0) {
  const int lane = threadIdx.x & 31;
  const bf16_t* p = base + (size_t)(row0 + (lane & 15)) * ldk + (k0 + ((lane >> 4) << 4));
  v8bf lo = *(const v8bf*)p;
  v8bf hi = *(const v8bf*)(p + 8);
  return __builtin_shufflevector(lo, hi, 0,1,2,3,4,5,6,7,8,9,10,11,12,13,14,15);
}

// B fragment: 32(K) x 16(N) from transposed storage Bt[N][K] (row-major, ld=ldk).
// lane<16: col=col0+lane, K=k0..k0+15 ; lane>=16: col-16, K=k0+16..k0+31
__device__ __forceinline__ v16bf load_b_frag(const bf16_t* base, int col0, int ldk, int k0) {
  const int lane = threadIdx.x & 31;
  const bf16_t* p = base + (size_t)(col0 + (lane & 15)) * ldk + (k0 + ((lane >> 4) << 4));
  v8bf lo = *(const v8bf*)p;
  v8bf hi = *(const v8bf*)(p + 8);
  return __builtin_shufflevector(lo, hi, 0,1,2,3,4,5,6,7,8,9,10,11,12,13,14,15);
}

// ---------------- async global->LDS copy (ASYNCcnt-tracked CDNA5 path) ----------------
__device__ __forceinline__ void async_copy16(const bf16_t* lds_dst, const bf16_t* gsrc) {
  unsigned ldsoff = (unsigned)(uintptr_t)lds_dst;             // LDS byte offset
  unsigned long long ga = (unsigned long long)(uintptr_t)gsrc;
  asm volatile("global_load_async_to_lds_b128 %0, %1, off"
               :: "v"(ldsoff), "v"(ga) : "memory");
}
__device__ __forceinline__ void wait_async8() {
  asm volatile("s_wait_asynccnt 0x8" ::: "memory");
}
__device__ __forceinline__ void wait_async0() {
  asm volatile("s_wait_asynccnt 0x0" ::: "memory");
}

// ---------------- shuffle reduction (wave32) ----------------
__device__ __forceinline__ float rsum32(float x) {
  #pragma unroll
  for (int m = 1; m < 32; m <<= 1) x += __shfl_xor(x, m, 32);
  return x;
}

// ---------------- setup kernels ----------------
__global__ __launch_bounds__(256) void init_ltn_kernel(const float* __restrict__ lat,
                                                       float* __restrict__ ltn) {
  size_t idx = (size_t)blockIdx.x * 256 + threadIdx.x;
  size_t per = (size_t)NLTN_ * D_;
  ltn[idx] = lat[idx % per];
}

// fp32 [L][K][N] -> bf16 transposed [L][N][K]
__global__ __launch_bounds__(256) void conv_transpose_kernel(const float* __restrict__ in,
                                                             bf16_t* __restrict__ out,
                                                             int K, int N) {
  size_t l = blockIdx.y;
  size_t idx = (size_t)blockIdx.x * 256 + threadIdx.x;
  size_t tot = (size_t)K * N;
  if (idx >= tot) return;
  int k = (int)(idx / N), n = (int)(idx % N);
  out[l * tot + (size_t)n * K + k] = f2bf(in[l * tot + idx]);
}

// ---------------- LayerNorm kernels (one wave per 768-row) ----------------
__global__ __launch_bounds__(256) void ln_ctx_kernel(
    const float* __restrict__ src, const float* __restrict__ pos,
    const float* __restrict__ ltn,
    const float* __restrict__ gs, const float* __restrict__ bs,
    const float* __restrict__ gl, const float* __restrict__ bl,
    bf16_t* __restrict__ ctx, bf16_t* __restrict__ tbuf) {
  const int row  = blockIdx.x * 8 + (threadIdx.x >> 5);
  const int lane = threadIdx.x & 31;
  const int g = row / NCTX_, j = row - g * NCTX_;
  const float* xp; const float* pp = nullptr;
  const float* gamma; const float* beta;
  bf16_t* tp = nullptr;
  if (j < NS_) {
    xp = src + ((size_t)g * NS_ + j) * D_;
    pp = pos + (size_t)(g & 3) * D_;                         // m = g % 4
    gamma = gs; beta = bs;
  } else {
    xp = ltn + ((size_t)g * NLTN_ + (j - NS_)) * D_;
    gamma = gl; beta = bl;
    tp = tbuf + ((size_t)g * NLTN_ + (j - NS_)) * D_;
  }
  float vals[24]; float s = 0.f;
  #pragma unroll
  for (int c = 0; c < 24; ++c) {
    int idx = lane + c * 32;
    float t = xp[idx];
    if (pp) t += pp[idx];
    vals[c] = t; s += t;
  }
  s = rsum32(s);
  float mu = s * (1.f / (float)D_);
  float vv = 0.f;
  #pragma unroll
  for (int c = 0; c < 24; ++c) { float d = vals[c] - mu; vv += d * d; }
  vv = rsum32(vv);
  float rstd = rsqrtf(vv * (1.f / (float)D_) + 1e-5f);
  bf16_t* op = ctx + (size_t)row * D_;
  #pragma unroll
  for (int c = 0; c < 24; ++c) {
    int idx = lane + c * 32;
    bf16_t yb = f2bf((vals[c] - mu) * rstd * gamma[idx] + beta[idx]);
    op[idx] = yb;
    if (tp) tp[idx] = yb;
  }
}

template<bool OUT_BF16>
__global__ __launch_bounds__(256) void ln_rows_kernel(
    const float* __restrict__ x, const float* __restrict__ gamma,
    const float* __restrict__ beta, void* __restrict__ out) {
  const int row  = blockIdx.x * 8 + (threadIdx.x >> 5);
  const int lane = threadIdx.x & 31;
  const float* xp = x + (size_t)row * D_;
  float vals[24]; float s = 0.f;
  #pragma unroll
  for (int c = 0; c < 24; ++c) { float t = xp[lane + c * 32]; vals[c] = t; s += t; }
  s = rsum32(s);
  float mu = s * (1.f / (float)D_);
  float vv = 0.f;
  #pragma unroll
  for (int c = 0; c < 24; ++c) { float d = vals[c] - mu; vv += d * d; }
  vv = rsum32(vv);
  float rstd = rsqrtf(vv * (1.f / (float)D_) + 1e-5f);
  #pragma unroll
  for (int c = 0; c < 24; ++c) {
    int idx = lane + c * 32;
    float y = (vals[c] - mu) * rstd * gamma[idx] + beta[idx];
    if (OUT_BF16) ((bf16_t*)out)[(size_t)row * D_ + idx] = f2bf(y);
    else          ((float*)out)[(size_t)row * D_ + idx]  = y;
  }
}

// ---------------- WMMA GEMM with async double-buffered LDS staging ----------------
// C[Mtot,N] = A[Mtot,K](bf16 row-major) * Bt[N,K](bf16 transposed-B)
// Block: 8 waves -> 128(M) x 128(N); K staged in 64-wide slabs
// (A:16KB + B:16KB per buffer, x2 buffers = 64KB LDS; 16 WMMAs/wave between barriers).
enum { EPI_BF16_SCALE = 0, EPI_RES_F32 = 1, EPI_GELU_BF16 = 2, EPI_KV = 3 };

template<int EPI>
__global__ __launch_bounds__(256) void gemm_wmma(
    const bf16_t* __restrict__ A, const bf16_t* __restrict__ Bt,
    int K, int N, void* out, const float* res, bf16_t* out2, float scale) {
  __shared__ __align__(16) bf16_t sA[2][128 * 64];
  __shared__ __align__(16) bf16_t sB[2][128 * 64];
  const int tid  = threadIdx.x;
  const int wave = tid >> 5;
  const int lane = tid & 31;
  const int m0 = blockIdx.y * 128;
  const int n0 = blockIdx.x * 128;

  // Stage one 64-K slab: 1024 chunks of 16B for each of A,B; 4 chunks per thread each.
  auto issue_stage = [&](int buf, int k0) {
    #pragma unroll
    for (int c = 0; c < 4; ++c) {
      const int chunk = tid + c * 256;         // 0..1023
      const int row   = chunk >> 3;            // 0..127
      const int kc    = (chunk & 7) * 8;       // element offset within slab
      async_copy16(&sA[buf][row * 64 + kc], A  + (size_t)(m0 + row) * K + k0 + kc);
      async_copy16(&sB[buf][row * 64 + kc], Bt + (size_t)(n0 + row) * K + k0 + kc);
    }
  };

  v8f acc[8];
  #pragma unroll
  for (int t = 0; t < 8; ++t) acc[t] = v8f{0.f,0.f,0.f,0.f,0.f,0.f,0.f,0.f};

  const int NK = K >> 6;
  issue_stage(0, 0);
  for (int it = 0; it < NK; ++it) {
    if (it + 1 < NK) { issue_stage((it + 1) & 1, (it + 1) * 64); wait_async8(); }
    else             { wait_async0(); }
    __syncthreads();                            // all waves' slab `it` visible
    const bf16_t* at = &sA[it & 1][0];
    const bf16_t* bt = &sB[it & 1][0];
    v16bf a0 = load_a_frag(at, wave * 16, 64, 0);
    v16bf a1 = load_a_frag(at, wave * 16, 64, 32);
    #pragma unroll
    for (int t = 0; t < 8; ++t) {
      v16bf b0 = load_b_frag(bt, t * 16, 64, 0);
      v16bf b1 = load_b_frag(bt, t * 16, 64, 32);
      acc[t] = wmma_bf16(a0, b0, acc[t]);
      acc[t] = wmma_bf16(a1, b1, acc[t]);
    }
    __syncthreads();                            // reads done before buffer reuse
  }

  // C/D layout: VGPR v -> row (lane<16 ? v : v+8), col = lane&15 of each 16-tile.
  const int mw    = m0 + wave * 16;
  const int rbase = (lane < 16) ? 0 : 8;
  const int ncol  = lane & 15;
  #pragma unroll
  for (int t = 0; t < 8; ++t) {
    #pragma unroll
    for (int v = 0; v < 8; ++v) {
      const int m = mw + rbase + v;
      const int n = n0 + t * 16 + ncol;
      float c = acc[t][v];
      if constexpr (EPI == EPI_BF16_SCALE) {
        ((bf16_t*)out)[(size_t)m * N + n] = f2bf(c * scale);
      } else if constexpr (EPI == EPI_RES_F32) {
        ((float*)out)[(size_t)m * N + n] = c + res[(size_t)m * N + n];
      } else if constexpr (EPI == EPI_GELU_BF16) {
        float gl = 0.5f * c * (1.f + erff(c * 0.70710678118f));
        ((bf16_t*)out)[(size_t)m * N + n] = f2bf(gl);
      } else {  // EPI_KV: cols [0,768) -> k rows ; [768,1536) -> v transposed per head
        const int gg = m / NCTX_, j = m - gg * NCTX_;
        if (n < E_) {
          ((bf16_t*)out)[(size_t)m * E_ + n] = f2bf(c);
        } else {
          const int d = n - E_, h = d >> 6, dd = d & 63;
          out2[(((size_t)(gg * H_ + h)) * DH_ + dd) * NCTX_ + j] = f2bf(c);
        }
      }
    }
  }
}

// ---------------- flash attention (transposed-score formulation) ----------------
// Block = (g, head), 4 waves x 16 query cols. q pre-scaled by DH^-0.5.
// sT[j][i] = sum_d k[j][d] q[i][d] : A = k rows, B = q rows (no transposes needed).
// oT[d][i] = sum_j vT[d][j] p[i][j]: A = vT rows, B = p (built in-register via xor-16).
// Softmax stats are per-lane (column i): ~2 shuffles per 32-j block; no LDS, no barriers.
__global__ __launch_bounds__(128) void attn_kernel(
    const bf16_t* __restrict__ q, const bf16_t* __restrict__ kk,
    const bf16_t* __restrict__ vt, bf16_t* __restrict__ o) {
  const int wave = threadIdx.x >> 5;
  const int lane = threadIdx.x & 31;
  const int g = blockIdx.x / H_, h = blockIdx.x % H_;
  const int i0 = wave * 16;

  const bf16_t* qbase = q  + (size_t)g * NLTN_ * E_ + h * DH_;       // ld E_
  const bf16_t* kbase = kk + (size_t)g * NCTX_ * E_ + h * DH_;       // ld E_
  const bf16_t* vbase = vt + ((size_t)(g * H_ + h)) * DH_ * NCTX_;   // ld NCTX_

  const v16bf qb0 = load_b_frag(qbase, i0, E_, 0);
  const v16bf qb1 = load_b_frag(qbase, i0, E_, 32);

  v8f oacc[4];
  #pragma unroll
  for (int t = 0; t < 4; ++t) oacc[t] = v8f{0.f,0.f,0.f,0.f,0.f,0.f,0.f,0.f};
  float mrun = -1e30f, lrun = 0.f;

  for (int j0 = 0; j0 < NCTX_; j0 += 32) {
    v8f s0 = v8f{0.f,0.f,0.f,0.f,0.f,0.f,0.f,0.f};
    v8f s1 = v8f{0.f,0.f,0.f,0.f,0.f,0.f,0.f,0.f};
    s0 = wmma_bf16(load_a_frag(kbase, j0,      E_, 0),  qb0, s0);
    s0 = wmma_bf16(load_a_frag(kbase, j0,      E_, 32), qb1, s0);
    s1 = wmma_bf16(load_a_frag(kbase, j0 + 16, E_, 0),  qb0, s1);
    s1 = wmma_bf16(load_a_frag(kbase, j0 + 16, E_, 32), qb1, s1);

    float mt = fmaxf(s0[0], s1[0]);
    #pragma unroll
    for (int v = 1; v < 8; ++v) mt = fmaxf(mt, fmaxf(s0[v], s1[v]));
    mt = fmaxf(mt, __shfl_xor(mt, 16, 32));
    const float mnew  = fmaxf(mrun, mt);
    const float alpha = __expf(mrun - mnew);
    mrun = mnew;
    float pa_f[8], pb_f[8]; float sum = 0.f;
    #pragma unroll
    for (int v = 0; v < 8; ++v) {
      pa_f[v] = __expf(s0[v] - mnew);
      pb_f[v] = __expf(s1[v] - mnew);
      sum += pa_f[v] + pb_f[v];
    }
    sum += __shfl_xor(sum, 16, 32);
    lrun = lrun * alpha + sum;
    #pragma unroll
    for (int t = 0; t < 4; ++t)
      #pragma unroll
      for (int v = 0; v < 8; ++v) oacc[t][v] *= alpha;

    // build B-fragment of P (Bt[i][j]) via xor-16 half-wave exchange
    unsigned pa[4], pb[4];
    #pragma unroll
    for (int u = 0; u < 4; ++u) {
      pa[u] = pk_bf16(pa_f[2 * u], pa_f[2 * u + 1]);   // j-tile a, consecutive j
      pb[u] = pk_bf16(pb_f[2 * u], pb_f[2 * u + 1]);   // j-tile b
    }
    v8u wv;
    #pragma unroll
    for (int u = 0; u < 4; ++u) {
      unsigned xa = (unsigned)__shfl_xor((int)pa[u], 16, 32);
      unsigned xb = (unsigned)__shfl_xor((int)pb[u], 16, 32);
      wv[u]     = (lane < 16) ? pa[u] : xb;            // K 0..7   of 32-slab
      wv[u + 4] = (lane < 16) ? xa    : pb[u];         // K 8..15 (/24..31)
    }
    const v16bf pf = __builtin_bit_cast(v16bf, wv);

    #pragma unroll
    for (int t = 0; t < 4; ++t)
      oacc[t] = wmma_bf16(load_a_frag(vbase, t * 16, NCTX_, j0), pf, oacc[t]);
  }

  // oT fragment: rows = d, cols = i; pack bf16 pairs -> b32 stores
  const float inv = 1.f / lrun;
  const int i     = i0 + (lane & 15);
  const int rbase = (lane < 16) ? 0 : 8;
  bf16_t* ob = o + ((size_t)g * NLTN_ + i) * E_ + h * DH_;
  #pragma unroll
  for (int t = 0; t < 4; ++t)
    #pragma unroll
    for (int u = 0; u < 4; ++u) {
      unsigned w = pk_bf16(oacc[t][2 * u] * inv, oacc[t][2 * u + 1] * inv);
      *(unsigned*)(ob + t * 16 + rbase + 2 * u) = w;
    }
}

// ---------------- host orchestration ----------------
static inline size_t alignUp256(size_t x) { return (x + 255) & ~(size_t)255; }

extern "C" void kernel_launch(void* const* d_in, const int* in_sizes, int n_in,
                              void* d_out, int out_size, void* d_ws, size_t ws_size,
                              hipStream_t stream) {
  (void)in_sizes; (void)n_in; (void)out_size; (void)ws_size;
  const float* src = (const float*)d_in[0];
  const float* lat = (const float*)d_in[1];
  const float* pos = (const float*)d_in[2];
  const float* wq  = (const float*)d_in[3];
  const float* wkv = (const float*)d_in[4];
  const float* wo  = (const float*)d_in[5];
  const float* w1  = (const float*)d_in[6];
  const float* w2  = (const float*)d_in[7];
  const float* lsg = (const float*)d_in[8];
  const float* lsb = (const float*)d_in[9];
  const float* llg = (const float*)d_in[10];
  const float* llb = (const float*)d_in[11];
  const float* lfg = (const float*)d_in[12];
  const float* lfb = (const float*)d_in[13];
  const float* ng  = (const float*)d_in[14];
  const float* nb  = (const float*)d_in[15];

  char* w = (char*)d_ws;
  auto carve = [&](size_t bytes) { char* p = w; w += alignUp256(bytes); return p; };
  float*  ltn   = (float*)carve((size_t)G_ * NLTN_ * D_ * 4);
  bf16_t* ctx   = (bf16_t*)carve((size_t)G_ * NCTX_ * D_ * 2);
  bf16_t* tbuf  = (bf16_t*)carve((size_t)G_ * NLTN_ * D_ * 2);
  bf16_t* qbuf  = (bf16_t*)carve((size_t)G_ * NLTN_ * E_ * 2);
  bf16_t* kbuf  = (bf16_t*)carve((size_t)G_ * NCTX_ * E_ * 2);
  bf16_t* vtbuf = (bf16_t*)carve((size_t)G_ * H_ * DH_ * NCTX_ * 2);
  bf16_t* attno = (bf16_t*)carve((size_t)G_ * NLTN_ * E_ * 2);
  bf16_t* hbuf  = (bf16_t*)carve((size_t)G_ * NLTN_ * D_ * 2);
  bf16_t* ffbuf = (bf16_t*)carve((size_t)G_ * NLTN_ * DFF_ * 2);
  bf16_t* wq_t  = (bf16_t*)carve((size_t)L_ * E_ * D_ * 2);
  bf16_t* wkv_t = (bf16_t*)carve((size_t)L_ * 2 * E_ * D_ * 2);
  bf16_t* wo_t  = (bf16_t*)carve((size_t)L_ * D_ * E_ * 2);
  bf16_t* w1_t  = (bf16_t*)carve((size_t)L_ * DFF_ * D_ * 2);
  bf16_t* w2_t  = (bf16_t*)carve((size_t)L_ * D_ * DFF_ * 2);

  auto tlaunch = [&](const float* in, bf16_t* outp, int K, int N) {
    size_t tot = (size_t)K * N;
    conv_transpose_kernel<<<dim3((unsigned)((tot + 255) / 256), L_), 256, 0, stream>>>(
        in, outp, K, N);
  };
  tlaunch(wq,  wq_t,  D_,   E_);
  tlaunch(wkv, wkv_t, D_,   2 * E_);
  tlaunch(wo,  wo_t,  E_,   D_);
  tlaunch(w1,  w1_t,  D_,   DFF_);
  tlaunch(w2,  w2_t,  DFF_, D_);

  init_ltn_kernel<<<dim3((G_ * NLTN_ * D_) / 256), 256, 0, stream>>>(lat, ltn);

  for (int l = 0; l < L_; ++l) {
    const bf16_t* wq_l  = wq_t  + (size_t)l * E_ * D_;
    const bf16_t* wkv_l = wkv_t + (size_t)l * 2 * E_ * D_;
    const bf16_t* wo_l  = wo_t  + (size_t)l * D_ * E_;
    const bf16_t* w1_l  = w1_t  + (size_t)l * DFF_ * D_;
    const bf16_t* w2_l  = w2_t  + (size_t)l * D_ * DFF_;

    ln_ctx_kernel<<<dim3(G_ * NCTX_ / 8), 256, 0, stream>>>(
        src, pos, ltn, lsg + l * D_, lsb + l * D_, llg + l * D_, llb + l * D_, ctx, tbuf);

    gemm_wmma<EPI_BF16_SCALE><<<dim3(E_ / 128, (G_ * NLTN_) / 128), 256, 0, stream>>>(
        tbuf, wq_l, D_, E_, qbuf, nullptr, nullptr, 0.125f);

    gemm_wmma<EPI_KV><<<dim3((2 * E_) / 128, (G_ * NCTX_) / 128), 256, 0, stream>>>(
        ctx, wkv_l, D_, 2 * E_, kbuf, nullptr, vtbuf, 1.f);

    attn_kernel<<<dim3(G_ * H_), 128, 0, stream>>>(qbuf, kbuf, vtbuf, attno);

    gemm_wmma<EPI_RES_F32><<<dim3(D_ / 128, (G_ * NLTN_) / 128), 256, 0, stream>>>(
        attno, wo_l, E_, D_, ltn, ltn, nullptr, 1.f);

    ln_rows_kernel<true><<<dim3(G_ * NLTN_ / 8), 256, 0, stream>>>(
        ltn, lfg + l * D_, lfb + l * D_, hbuf);

    gemm_wmma<EPI_GELU_BF16><<<dim3(DFF_ / 128, (G_ * NLTN_) / 128), 256, 0, stream>>>(
        hbuf, w1_l, D_, DFF_, ffbuf, nullptr, nullptr, 1.f);

    gemm_wmma<EPI_RES_F32><<<dim3(D_ / 128, (G_ * NLTN_) / 128), 256, 0, stream>>>(
        ffbuf, w2_l, DFF_, D_, ltn, ltn, nullptr, 1.f);
  }

  ln_rows_kernel<false><<<dim3(G_ * NLTN_ / 8), 256, 0, stream>>>(ltn, ng, nb, d_out);
}
